// AttentionDecoder_26439818674419
// MI455X (gfx1250) — matrix operand order repaired and checked by reading, not compiled, not run on hardware
//
#include <hip/hip_runtime.h>
#include <hip/hip_bf16.h>
#include <stdint.h>

#define L_LAYERS 2
#define BB 32
#define TT 64
#define SS 128
#define HH 1024
#define VV 32000
#define GG (4 * HH)

typedef __bf16 v16bf __attribute__((ext_vector_type(16)));
typedef float v8f __attribute__((ext_vector_type(8)));
typedef unsigned int v4u __attribute__((ext_vector_type(4)));
typedef int v8i __attribute__((ext_vector_type(8)));
typedef int v4i __attribute__((ext_vector_type(4)));

union FragBF16 {
  v16bf v;
  v4u q[2];
};

static __device__ __forceinline__ unsigned short f32_to_bf16(float f) {
  union { float f; unsigned u; } a;
  a.f = f;
  unsigned u = a.u;
  unsigned r = u + 0x7FFFu + ((u >> 16) & 1u);
  if ((u & 0x7F800000u) == 0x7F800000u) r = u;  // inf/nan passthrough
  return (unsigned short)(r >> 16);
}

static __device__ __forceinline__ float sigmoidf_fast(float x) {
  return 1.0f / (1.0f + __expf(-x));
}

static __device__ __forceinline__ unsigned lds_off(const void* p) {
  // Generic LDS-aperture address: low 32 bits are the wave-relative LDS byte offset.
  return (unsigned)(unsigned long long)p;
}

static __device__ __forceinline__ v8f wmma_bf16(v16bf a, v16bf b, v8f c) {
  return __builtin_amdgcn_wmma_f32_16x16x32_bf16(false, a, false, b, (short)0, c,
                                                 false, false);
}

// TDM: load a tile_rows x tile_cols tile of 2-byte elements (row stride
// row_stride_elems) from global into LDS, packed row-major [rows][cols].
// Tensor dims == tile dims (all shapes divide exactly; no OOB clipping needed).
static __device__ __forceinline__ void tdm_load_tile_bf16(
    unsigned lds_byte_addr, const void* gptr, unsigned tile_cols,
    unsigned tile_rows, unsigned long long row_stride_elems) {
  unsigned long long ga = (unsigned long long)gptr;
  v4u g0;
  g0[0] = 1u;                                          // count=1, user D#
  g0[1] = lds_byte_addr;                               // lds_addr
  g0[2] = (unsigned)(ga & 0xFFFFFFFFull);              // global_addr[31:0]
  g0[3] = (unsigned)((ga >> 32) & 0x1FFFFFFull) | 0x80000000u;  // addr[56:32] | type=2
  v8i g1;
  g1[0] = (int)(1u << 16);                             // data_size=1 (2 bytes)
  g1[1] = (int)((tile_cols & 0xFFFFu) << 16);          // tensor_dim0 lo
  g1[2] = (int)(((tile_cols >> 16) & 0xFFFFu) | ((tile_rows & 0xFFFFu) << 16));
  g1[3] = (int)(((tile_rows >> 16) & 0xFFFFu) | ((tile_cols & 0xFFFFu) << 16)); // tile_dim0
  g1[4] = (int)(tile_rows & 0xFFFFu);                  // tile_dim1, tile_dim2=0
  g1[5] = (int)(row_stride_elems & 0xFFFFFFFFull);     // tensor_dim0_stride lo
  g1[6] = (int)((row_stride_elems >> 32) & 0xFFFFull); // stride hi, dim1_stride=0
  g1[7] = 0;
  v4i g2 = {0, 0, 0, 0};
  v4i g3 = {0, 0, 0, 0};
#if __clang_major__ >= 23
  v8i g4 = {0, 0, 0, 0, 0, 0, 0, 0};
  __builtin_amdgcn_tensor_load_to_lds(g0, g1, g2, g3, g4, 0);
#else
  __builtin_amdgcn_tensor_load_to_lds(g0, g1, g2, g3, 0);
#endif
}

// A-fragment (16x32 bf16) from an LDS tile laid out [rows][ldk] row-major.
// Contiguous K order used by this code: lane holds row M = lane%16; K 16*s..16*s+15, s=lane/16.
static __device__ __forceinline__ v16bf load_a_frag(const unsigned short* lds,
                                                    int row_base, int ldk) {
  const int lane = threadIdx.x & 31;
  const int row = row_base + (lane & 15);
  const int s = lane >> 4;
  FragBF16 f;
  f.q[0] = *(const v4u*)(lds + row * ldk + s * 16);
  f.q[1] = *(const v4u*)(lds + row * ldk + s * 16 + 8);
  return f.v;
}

// B-fragment (32x16 bf16) from an LDS tile laid out [cols][ldk] (each B column
// = one contiguous K row). Contiguous K order used here: lane = col N = lane%16, K 16*(lane/16)..+15.
static __device__ __forceinline__ v16bf load_b_frag_lds(const unsigned short* lds,
                                                        int col_base, int ldk) {
  const int lane = threadIdx.x & 31;
  const int col = col_base + (lane & 15);
  const int s = lane >> 4;
  FragBF16 f;
  f.q[0] = *(const v4u*)(lds + col * ldk + s * 16);
  f.q[1] = *(const v4u*)(lds + col * ldk + s * 16 + 8);
  return f.v;
}

// B-fragment gathered straight from a row-major [rows, HH] weight matrix
// (column n of B = row n of W, contiguous in K).
static __device__ __forceinline__ v16bf load_b_frag_gmem(
    const unsigned short* __restrict__ W, int row_base, int kidx) {
  const int lane = threadIdx.x & 31;
  const unsigned short* p =
      W + (long long)(row_base + (lane & 15)) * HH + kidx + ((lane >> 4) << 4);
  FragBF16 f;
  f.q[0] = *(const v4u*)(p);
  f.q[1] = *(const v4u*)(p + 8);
  return f.v;
}

__global__ void k_cvt_bf16(unsigned short* __restrict__ dst,
                           const float* __restrict__ src, int n) {
  for (int i = blockIdx.x * blockDim.x + threadIdx.x; i < n;
       i += gridDim.x * blockDim.x)
    dst[i] = f32_to_bf16(src[i]);
}

__global__ void k_copy_f32(float* __restrict__ dst, const float* __restrict__ src,
                           int n) {
  for (int i = blockIdx.x * blockDim.x + threadIdx.x; i < n;
       i += gridDim.x * blockDim.x)
    dst[i] = src[i];
}

__global__ void k_embed(const int* __restrict__ idx, const float* __restrict__ emb,
                        float* __restrict__ xf, unsigned short* __restrict__ xq,
                        int n) {
  for (int i = blockIdx.x * blockDim.x + threadIdx.x; i < n;
       i += gridDim.x * blockDim.x) {
    const int bt = i >> 10;  // / HH
    const int h = i & (HH - 1);
    const int e = idx[bt];
    float v = emb[(long long)e * HH + h];
    v = v > 0.0f ? v : 0.0f;  // relu (padding row is already zero)
    xf[i] = v;
    xq[i] = f32_to_bf16(v);
  }
}

// One LSTM layer for one timestep: gates[32,4096] = A@Wih^T + h@Whh^T + biases,
// fused with gate nonlinearities, cell/hidden update and residual add.
// Each wave owns 16 hidden units across all 4 gates -> i,f,g,o for the same
// (b, n) land in the same lane/register slot.
__global__ void __launch_bounds__(256) k_lstm_layer(
    const unsigned short* __restrict__ Abase, long long a_stride,
    const unsigned short* __restrict__ Hin, const unsigned short* __restrict__ Wih,
    const unsigned short* __restrict__ Whh, const float* __restrict__ bih,
    const float* __restrict__ bhh, const float* __restrict__ resid,
    long long r_stride, float* __restrict__ c_st, float* __restrict__ h_f32,
    unsigned short* __restrict__ h_q, float* __restrict__ out_f32,
    unsigned short* __restrict__ out_q, unsigned short* __restrict__ outs_slot,
    long long outs_stride) {
  __shared__ __align__(16) unsigned short ldsA[2][2][32 * 32];  // [buf][inp|h]
  const int tid = threadIdx.x;
  const int wv = tid >> 5;
  const int lane = tid & 31;
  const int n_w = blockIdx.x * 128 + wv * 16;  // 16 hidden units per wave

  v8f acc[2][4];
#pragma unroll
  for (int m = 0; m < 2; ++m)
#pragma unroll
    for (int g = 0; g < 4; ++g)
#pragma unroll
      for (int e = 0; e < 8; ++e) acc[m][g][e] = 0.0f;

  if (wv == 0) {
    tdm_load_tile_bf16(lds_off(&ldsA[0][0][0]), Abase, 32, 32,
                       (unsigned long long)a_stride);
    tdm_load_tile_bf16(lds_off(&ldsA[0][1][0]), Hin, 32, 32,
                       (unsigned long long)HH);
  }

  for (int kk = 0; kk < HH / 32; ++kk) {
    if (wv == 0) __builtin_amdgcn_s_wait_tensorcnt(0);
    __syncthreads();
    if (wv == 0 && kk + 1 < HH / 32) {
      const int nb = (kk + 1) & 1;
      tdm_load_tile_bf16(lds_off(&ldsA[nb][0][0]), Abase + (kk + 1) * 32, 32, 32,
                         (unsigned long long)a_stride);
      tdm_load_tile_bf16(lds_off(&ldsA[nb][1][0]), Hin + (kk + 1) * 32, 32, 32,
                         (unsigned long long)HH);
    }
    const int cb = kk & 1;
    const unsigned short* tA = &ldsA[cb][0][0];
    const unsigned short* tH = &ldsA[cb][1][0];
    v16bf a0 = load_a_frag(tA, 0, 32);
    v16bf a1 = load_a_frag(tA, 16, 32);
    v16bf hh0 = load_a_frag(tH, 0, 32);
    v16bf hh1 = load_a_frag(tH, 16, 32);
    const int kidx = kk * 32;
    if (kk + 1 < HH / 32) {  // warm next K slab of weights (L2-resident)
      __builtin_prefetch(Wih + (long long)(n_w + (lane & 15)) * HH + kidx + 32, 0, 0);
      __builtin_prefetch(Whh + (long long)(n_w + (lane & 15)) * HH + kidx + 32, 0, 0);
    }
#pragma unroll
    for (int g = 0; g < 4; ++g) {
      v16bf bi = load_b_frag_gmem(Wih, g * HH + n_w, kidx);
      v16bf bh = load_b_frag_gmem(Whh, g * HH + n_w, kidx);
      acc[0][g] = wmma_bf16(a0, bi, acc[0][g]);
      acc[1][g] = wmma_bf16(a1, bi, acc[1][g]);
      acc[0][g] = wmma_bf16(hh0, bh, acc[0][g]);
      acc[1][g] = wmma_bf16(hh1, bh, acc[1][g]);
    }
  }

  // Epilogue: C/D layout -> lane holds n = n_w + lane%16, rows r + 8*(lane/16).
  const int n = n_w + (lane & 15);
  const int half8 = (lane >> 4) * 8;
  float bias[4];
#pragma unroll
  for (int g = 0; g < 4; ++g) bias[g] = bih[g * HH + n] + bhh[g * HH + n];
#pragma unroll
  for (int m = 0; m < 2; ++m) {
#pragma unroll
    for (int r = 0; r < 8; ++r) {
      const int b = m * 16 + half8 + r;
      const long long idx = (long long)b * HH + n;
      const float ig = acc[m][0][r] + bias[0];
      const float fg = acc[m][1][r] + bias[1];
      const float gg = acc[m][2][r] + bias[2];
      const float og = acc[m][3][r] + bias[3];
      const float cn =
          sigmoidf_fast(fg) * c_st[idx] + sigmoidf_fast(ig) * tanhf(gg);
      const float hn = sigmoidf_fast(og) * tanhf(cn);
      const float on = hn + resid[(long long)b * r_stride + n];
      c_st[idx] = cn;
      h_f32[idx] = hn;
      h_q[idx] = f32_to_bf16(hn);
      out_f32[idx] = on;
      out_q[idx] = f32_to_bf16(on);
      if (outs_slot) outs_slot[(long long)b * outs_stride + n] = f32_to_bf16(on);
    }
  }
}

// Dot attention + softmax for one step: scores[b,s] = enc[b,s,:].last[b,:]
__global__ void __launch_bounds__(256) k_attn(const float* __restrict__ enc,
                                              const float* __restrict__ last,
                                              float* __restrict__ attn_out,
                                              int t) {
  __shared__ float sc[SS];
  __shared__ float tmp[SS];
  const int b = blockIdx.x;
  const int tid = threadIdx.x;
  const int wv = tid >> 5;
  const int lane = tid & 31;
  for (int s = wv; s < SS; s += 8) {
    const float* er = enc + ((long long)b * SS + s) * HH;
    const float* lr = last + (long long)b * HH;
    float p = 0.0f;
    for (int h = lane; h < HH; h += 32) p += er[h] * lr[h];
#pragma unroll
    for (int off = 16; off; off >>= 1) p += __shfl_xor(p, off, 32);
    if (lane == 0) sc[s] = p;
  }
  __syncthreads();
  if (tid < SS) tmp[tid] = sc[tid];
  __syncthreads();
  for (int off = SS / 2; off >= 1; off >>= 1) {
    if (tid < off) tmp[tid] = fmaxf(tmp[tid], tmp[tid + off]);
    __syncthreads();
  }
  const float mx = tmp[0];
  __syncthreads();
  float e = 0.0f;
  if (tid < SS) {
    e = __expf(sc[tid] - mx);
    tmp[tid] = e;
  }
  __syncthreads();
  for (int off = SS / 2; off >= 1; off >>= 1) {
    if (tid < off) tmp[tid] += tmp[tid + off];
    __syncthreads();
  }
  if (tid < SS)
    attn_out[((long long)b * SS + tid) * TT + t] = e / tmp[0];
}

// Classifier GEMM: [2048,1024]bf16 x [1024,32000]bf16 -> f32 logits + bias.
// Block tile 128x128, 8 waves of 64x32, TDM double-buffered A/B tiles.
__global__ void __launch_bounds__(256) k_classifier(
    const unsigned short* __restrict__ Aq, const unsigned short* __restrict__ Bq,
    const float* __restrict__ bias, float* __restrict__ logits) {
  __shared__ __align__(16) unsigned short ldsA[2][128 * 32];
  __shared__ __align__(16) unsigned short ldsB[2][128 * 32];
  const int tid = threadIdx.x;
  const int wv = tid >> 5;
  const int lane = tid & 31;
  const int m0 = blockIdx.y * 128;
  const int v0 = blockIdx.x * 128;
  const int wm = wv >> 2;
  const int wn = wv & 3;

  v8f acc[4][2];
#pragma unroll
  for (int i = 0; i < 4; ++i)
#pragma unroll
    for (int j = 0; j < 2; ++j)
#pragma unroll
      for (int e = 0; e < 8; ++e) acc[i][j][e] = 0.0f;

  const unsigned short* Ab = Aq + (long long)m0 * HH;
  const unsigned short* Bb = Bq + (long long)v0 * HH;

  if (wv == 0) {
    tdm_load_tile_bf16(lds_off(&ldsA[0][0]), Ab, 32, 128, HH);
    tdm_load_tile_bf16(lds_off(&ldsB[0][0]), Bb, 32, 128, HH);
  }
  for (int kk = 0; kk < HH / 32; ++kk) {
    if (wv == 0) __builtin_amdgcn_s_wait_tensorcnt(0);
    __syncthreads();
    if (wv == 0 && kk + 1 < HH / 32) {
      const int nb = (kk + 1) & 1;
      tdm_load_tile_bf16(lds_off(&ldsA[nb][0]), Ab + (kk + 1) * 32, 32, 128, HH);
      tdm_load_tile_bf16(lds_off(&ldsB[nb][0]), Bb + (kk + 1) * 32, 32, 128, HH);
    }
    const int cb = kk & 1;
    v16bf a[4], bt[2];
#pragma unroll
    for (int i = 0; i < 4; ++i)
      a[i] = load_a_frag(&ldsA[cb][0], wm * 64 + i * 16, 32);
#pragma unroll
    for (int j = 0; j < 2; ++j)
      bt[j] = load_b_frag_lds(&ldsB[cb][0], wn * 32 + j * 16, 32);
#pragma unroll
    for (int i = 0; i < 4; ++i)
#pragma unroll
      for (int j = 0; j < 2; ++j) acc[i][j] = wmma_bf16(a[i], bt[j], acc[i][j]);
  }
  const int half8 = (lane >> 4) * 8;
#pragma unroll
  for (int i = 0; i < 4; ++i) {
#pragma unroll
    for (int j = 0; j < 2; ++j) {
      const int vcol = v0 + wn * 32 + j * 16 + (lane & 15);
      const float bv = bias[vcol];
#pragma unroll
      for (int r = 0; r < 8; ++r) {
        const int mrow = m0 + wm * 64 + i * 16 + half8 + r;
        logits[(long long)mrow * VV + vcol] = acc[i][j][r] + bv;
      }
    }
  }
}

extern "C" void kernel_launch(void* const* d_in, const int* in_sizes, int n_in,
                              void* d_out, int out_size, void* d_ws,
                              size_t ws_size, hipStream_t stream) {
  (void)in_sizes; (void)n_in; (void)out_size; (void)ws_size;
  const int* dec = (const int*)d_in[0];
  const float* h0 = (const float*)d_in[1];
  const float* c0 = (const float*)d_in[2];
  const float* enc = (const float*)d_in[3];
  const float* emb = (const float*)d_in[4];
  const float* Wih = (const float*)d_in[5];
  const float* Whh = (const float*)d_in[6];
  const float* bih = (const float*)d_in[7];
  const float* bhh = (const float*)d_in[8];
  const float* clsW = (const float*)d_in[9];
  const float* clsb = (const float*)d_in[10];
  float* out = (float*)d_out;

  char* ws = (char*)d_ws;
  size_t cur = 0;
  auto alloc = [&](size_t bytes) -> char* {
    char* p = ws + cur;
    cur += (bytes + 255) & ~(size_t)255;
    return p;
  };
  unsigned short* xq = (unsigned short*)alloc((size_t)BB * TT * HH * 2);
  float* xf = (float*)alloc((size_t)BB * TT * HH * 4);
  unsigned short* wihq = (unsigned short*)alloc((size_t)L_LAYERS * GG * HH * 2);
  unsigned short* whhq = (unsigned short*)alloc((size_t)L_LAYERS * GG * HH * 2);
  unsigned short* clsq = (unsigned short*)alloc((size_t)VV * HH * 2);
  unsigned short* hq0 = (unsigned short*)alloc((size_t)L_LAYERS * BB * HH * 2);
  unsigned short* hq1 = (unsigned short*)alloc((size_t)L_LAYERS * BB * HH * 2);
  float* hF = (float*)alloc((size_t)L_LAYERS * BB * HH * 4);
  float* cF = (float*)alloc((size_t)L_LAYERS * BB * HH * 4);
  float* inpAf = (float*)alloc((size_t)BB * HH * 4);
  unsigned short* inpAq = (unsigned short*)alloc((size_t)BB * HH * 2);
  float* inpBf = (float*)alloc((size_t)BB * HH * 4);
  unsigned short* inpBq = (unsigned short*)alloc((size_t)BB * HH * 2);
  unsigned short* outsq = (unsigned short*)alloc((size_t)BB * TT * HH * 2);

  const size_t LOGITS_N = (size_t)BB * TT * VV;
  float* out_hT = out + LOGITS_N;
  float* out_cT = out_hT + (size_t)L_LAYERS * BB * HH;
  float* out_attn = out_cT + (size_t)L_LAYERS * BB * HH;

  // bf16 weight/state preparation (weights stay L2-resident afterwards)
  k_cvt_bf16<<<2048, 256, 0, stream>>>(wihq, Wih, L_LAYERS * GG * HH);
  k_cvt_bf16<<<2048, 256, 0, stream>>>(whhq, Whh, L_LAYERS * GG * HH);
  k_cvt_bf16<<<4096, 256, 0, stream>>>(clsq, clsW, VV * HH);
  k_cvt_bf16<<<64, 256, 0, stream>>>(hq0, h0, L_LAYERS * BB * HH);
  k_copy_f32<<<64, 256, 0, stream>>>(cF, c0, L_LAYERS * BB * HH);
  k_embed<<<1024, 256, 0, stream>>>(dec, emb, xf, xq, BB * TT * HH);

  for (int t = 0; t < TT; ++t) {
    const unsigned short* hin = (t & 1) ? hq1 : hq0;
    unsigned short* hout = (t & 1) ? hq0 : hq1;
    // layer 0: A = x[:,t,:], residual = x[:,t,:]
    k_lstm_layer<<<8, 256, 0, stream>>>(
        xq + (size_t)t * HH, (long long)TT * HH, hin, wihq, whhq, bih, bhh,
        xf + (size_t)t * HH, (long long)TT * HH, cF, hF, hout, inpAf, inpAq,
        (unsigned short*)nullptr, 0);
    // layer 1: A = inpA, residual = inpA; also emits outs[:,t,:] for classifier
    k_lstm_layer<<<8, 256, 0, stream>>>(
        inpAq, (long long)HH, hin + (size_t)BB * HH, wihq + (size_t)GG * HH,
        whhq + (size_t)GG * HH, bih + GG, bhh + GG, inpAf, (long long)HH,
        cF + (size_t)BB * HH, hF + (size_t)BB * HH, hout + (size_t)BB * HH,
        inpBf, inpBq, outsq + (size_t)t * HH, (long long)TT * HH);
    k_attn<<<BB, 256, 0, stream>>>(enc, inpBf, out_attn, t);
  }

  k_classifier<<<dim3(VV / 128, (BB * TT) / 128), 256, 0, stream>>>(outsq, clsq,
                                                                   clsb, out);
  k_copy_f32<<<64, 256, 0, stream>>>(out_hT, hF, L_LAYERS * BB * HH);
  k_copy_f32<<<64, 256, 0, stream>>>(out_cT, cF, L_LAYERS * BB * HH);
}
